// VGG_RP_91122026152553
// MI455X (gfx1250) — matrix operand order repaired and checked by reading, not compiled, not run on hardware
//
#include <hip/hip_runtime.h>
#include <hip/hip_bf16.h>

typedef __attribute__((ext_vector_type(16))) _Float16 v16h;
typedef __attribute__((ext_vector_type(8)))  _Float16 v8h;
typedef __attribute__((ext_vector_type(8)))  float    v8f;

static __device__ __forceinline__ v8f wmma_f16(v16h a, v16h b, v8f c) {
  // 8 args: (neg_a, A, neg_b, B, c_mod, C, reuse_a, reuse_b)
  return __builtin_amdgcn_wmma_f32_16x16x32_f16(false, a, false, b, (short)0, c, false, false);
}

// ------------------------------------------------------------- TDM A-staging
#if __has_builtin(__builtin_amdgcn_tensor_load_to_lds)
#define USE_TDM 1
typedef unsigned int tdm_u32x4 __attribute__((ext_vector_type(4)));
typedef int          tdm_i32x4 __attribute__((ext_vector_type(4)));
typedef int          tdm_i32x8 __attribute__((ext_vector_type(8)));

// 2D tile load: `rows` rows of `tileK` f16 elements, global row stride = strideElems.
static __device__ __forceinline__ void tdm_load_tile_2d(const void* g, unsigned lds,
                                                        unsigned tileK, unsigned rows,
                                                        unsigned strideElems) {
  unsigned long long ga = (unsigned long long)(uintptr_t)g;
  unsigned dim0 = strideElems;     // tensor_dim0 (elements)
  unsigned dim1 = 1u << 20;        // generous row count for OOB logic
  tdm_u32x4 g0;
  g0[0] = 1u;                                                    // count=1 (valid D#)
  g0[1] = lds;                                                   // lds_addr
  g0[2] = (unsigned)ga;                                          // global_addr[31:0]
  g0[3] = (unsigned)((ga >> 32) & 0x01ffffffu) | (2u << 30);     // addr[56:32] | type=2
  tdm_i32x8 g1;
  g1[0] = (int)(1u << 16);                                       // data_size=1 (2 bytes)
  g1[1] = (int)((dim0 & 0xffffu) << 16);                         // tensor_dim0 lo
  g1[2] = (int)((dim0 >> 16) | ((dim1 & 0xffffu) << 16));        // dim0 hi | dim1 lo
  g1[3] = (int)((dim1 >> 16) | (tileK << 16));                   // dim1 hi | tile_dim0
  g1[4] = (int)rows;                                             // tile_dim1 (tile_dim2=0)
  g1[5] = (int)strideElems;                                      // tensor_dim0_stride lo
  g1[6] = 0;
  g1[7] = 0;
  tdm_i32x4 z4 = {0, 0, 0, 0};
#if defined(__clang_major__) && (__clang_major__ >= 23)
  tdm_i32x8 z8 = {0, 0, 0, 0, 0, 0, 0, 0};
  __builtin_amdgcn_tensor_load_to_lds(g0, g1, z4, z4, z8, 0);
#else
  __builtin_amdgcn_tensor_load_to_lds(g0, g1, z4, z4, 0);
#endif
}
#else
#define USE_TDM 0
#endif

// ---------------------------------------------------------------- conversions
// NCHW f32 -> NHWC f16
__global__ __launch_bounds__(256) void nchw_to_nhwc_f16_kernel(const float* __restrict__ in,
                                                               _Float16* __restrict__ out,
                                                               int B, int C, int HW) {
  int i = blockIdx.x * blockDim.x + threadIdx.x;
  if (i >= B * C * HW) return;
  int b = i / (C * HW), rem = i - b * C * HW;
  int c = rem / HW, p = rem - c * HW;
  out[((size_t)b * HW + p) * C + c] = (_Float16)in[i];
}

// layer0: OIHW f32 [64,3,3,3] -> f16 [64, 32] with k = ci*9 + kh*3 + kw (zero-padded)
__global__ __launch_bounds__(256) void pack_weights_kernel(const float* __restrict__ w,
                                                           _Float16* __restrict__ o,
                                                           int cout, int cink, int kp) {
  int i = blockIdx.x * blockDim.x + threadIdx.x;
  if (i >= cout * kp) return;
  int oc = i / kp, k = i - oc * kp;
  o[i] = (k < cink) ? (_Float16)w[oc * cink + k] : (_Float16)0.f;
}

// channels-last layers: OIHW f32 -> f16 [Cout, 9*Cin] with k = (kh*3+kw)*Cin + ci
__global__ __launch_bounds__(256) void pack_weights_cl_kernel(const float* __restrict__ w,
                                                              _Float16* __restrict__ o,
                                                              int cout, int cin) {
  int i = blockIdx.x * blockDim.x + threadIdx.x;
  int K9 = cin * 9;
  if (i >= cout * K9) return;
  int oc = i / K9, k = i - oc * K9;
  int r = k / cin, ci = k - r * cin;
  o[i] = (_Float16)w[(size_t)oc * K9 + ci * 9 + r];
}

// Stack cls(3x512)+reg(12x512) into one 16x512 f16 A matrix + 16 f32 biases
__global__ __launch_bounds__(256) void pack_heads_kernel(const float* __restrict__ cls_w,
                                                         const float* __restrict__ cls_b,
                                                         const float* __restrict__ reg_w,
                                                         const float* __restrict__ reg_b,
                                                         _Float16* __restrict__ hw,
                                                         float* __restrict__ hb) {
  int i = blockIdx.x * blockDim.x + threadIdx.x;
  if (i >= 16 * 512) return;
  int row = i >> 9, k = i & 511;
  float v = 0.f;
  if (row < 3)       v = cls_w[row * 512 + k];
  else if (row < 15) v = reg_w[(row - 3) * 512 + k];
  hw[i] = (_Float16)v;
  if (i < 16) hb[i] = (i < 3) ? cls_b[i] : (i < 15 ? reg_b[i - 3] : 0.f);
}

// ------------------------------------------------ layer0: LDS-gather (Cin=3)
__global__ __launch_bounds__(128) void conv0_wmma_kernel(const _Float16* __restrict__ act,
                                                         const _Float16* __restrict__ wgt,
                                                         const float* __restrict__ bias,
                                                         _Float16* __restrict__ out, int B) {
  constexpr int CIN = 3, COUT = 64, H = 224, W = 224, HO = 112, WO = 112, NP = HO * WO;
  __shared__ __align__(64) _Float16 Bs[64][32];
  const int tid = threadIdx.x, lane = tid & 31, wave = tid >> 5;
  const int lr = lane & 15, lh = lane >> 4;
  const int mBase = wave * 16;
  const int b = blockIdx.y;

  const int col = tid & 63, kHalf = tid >> 6;
  int sub = col >> 4, n = col & 15;
  int p = blockIdx.x * 16 + n;
  bool cvalid = p < NP;
  int pp = cvalid ? p : 0;
  int poY = pp / WO, poX = pp - poY * WO;
  int oy = poY * 2 + (sub >> 1), ox = poX * 2 + (sub & 1);
  const _Float16* cbase = act + (size_t)b * H * W * CIN;

  #pragma unroll
  for (int e = 0; e < 16; ++e) {
    int kk = kHalf * 16 + e;
    _Float16 v = (_Float16)0.f;
    if (cvalid && kk < 27) {
      int ci = kk / 9, r = kk - ci * 9;
      int kh = r / 3, kw = r - kh * 3;
      int yy = oy + kh - 1, xx = ox + kw - 1;
      if (yy >= 0 && yy < H && xx >= 0 && xx < W)
        v = cbase[((size_t)yy * W + xx) * CIN + ci];
    }
    Bs[col][kHalf * 16 + e] = v;
  }
  __syncthreads();
  const _Float16* arow = wgt + (size_t)(mBase + lr) * 32 + lh * 16;
  union { v16h v; v8h h[2]; } A;
  A.h[0] = *(const v8h*)(arow);
  A.h[1] = *(const v8h*)(arow + 8);
  v16h b0 = *(const v16h*)&Bs[ 0 + lr][lh * 16];
  v16h b1 = *(const v16h*)&Bs[16 + lr][lh * 16];
  v16h b2 = *(const v16h*)&Bs[32 + lr][lh * 16];
  v16h b3 = *(const v16h*)&Bs[48 + lr][lh * 16];
  v8f a0 = {}, a1 = {}, a2 = {}, a3 = {};
  a0 = wmma_f16(A.v, b0, a0);
  a1 = wmma_f16(A.v, b1, a1);
  a2 = wmma_f16(A.v, b2, a2);
  a3 = wmma_f16(A.v, b3, a3);

  int q = blockIdx.x * 16 + lr;
  if (q < NP) {
    int pY = q / WO, pX = q - pY * WO;
    v8h o8;
    #pragma unroll
    for (int j = 0; j < 8; ++j) {
      int co = mBase + lh * 8 + j;
      float v = fmaxf(fmaxf(a0[j], a1[j]), fmaxf(a2[j], a3[j])) + bias[co];
      o8[j] = (_Float16)fmaxf(v, 0.f);
    }
    *(v8h*)&out[(((size_t)b * HO + pY) * WO + pX) * COUT + mBase + lh * 8] = o8;
  }
}

// ----------------- channels-last implicit-GEMM conv: direct B fragments + TDM A
// K ordered (kh,kw)-major: k = r*CIN + ci  =>  a lane's 16 B K-values are 32
// contiguous bytes of NHWC activations. A (64x32 block tile) staged by a
// double-buffered, software-pipelined tensor_load_to_lds (one barrier/step).
template<int CIN, int COUT, int H, int W, int PAD, bool POOL>
__global__ __launch_bounds__(128) void conv_cl_wmma_kernel(const _Float16* __restrict__ act,
                                                           const _Float16* __restrict__ wgt,
                                                           const float* __restrict__ bias,
                                                           _Float16* __restrict__ out, int B) {
  constexpr int K9 = 9 * CIN;
  constexpr int HO = POOL ? (H / 2) : (H - 2);
  constexpr int WO = POOL ? (W / 2) : (W - 2);
  constexpr int NP = HO * WO;

  __shared__ __align__(16) _Float16 As[2][64][32];   // double-buffered A tile

  const int tid  = threadIdx.x, lane = tid & 31, wave = tid >> 5;
  const int lr = lane & 15, lh = lane >> 4;
  const int mRow  = wave * 16 + lr;       // A row within block tile
  const int mBase = blockIdx.z * 64;

  int oy[4], ox[4];
  const _Float16* base[4];
  bool cv[4];
  #pragma unroll
  for (int t = 0; t < 4; ++t) {
    if constexpr (POOL) {
      int b = blockIdx.y;
      int p = blockIdx.x * 16 + lr;
      cv[t] = p < NP;
      int pp = cv[t] ? p : 0;
      int poY = pp / WO, poX = pp - poY * WO;
      oy[t] = poY * 2 + (t >> 1);
      ox[t] = poX * 2 + (t & 1);
      base[t] = act + (size_t)b * H * W * CIN;
    } else {
      int g = blockIdx.x * 64 + t * 16 + lr;
      cv[t] = g < B * NP;
      int gg = cv[t] ? g : 0;
      int cb = gg / NP, pix = gg - cb * NP;
      int oyt = pix / WO;
      oy[t] = oyt;
      ox[t] = pix - oyt * WO;
      base[t] = act + (size_t)cb * H * W * CIN;
    }
  }

  v8f acc0 = {}, acc1 = {}, acc2 = {}, acc3 = {};
  const unsigned ldsA = (unsigned)(uintptr_t)&As[0][0][0];
  const _Float16* wblk = wgt + (size_t)mBase * K9;
  int buf = 0;

#if USE_TDM
  if (wave == 0) tdm_load_tile_2d(wblk, ldsA, 32, 64, K9);   // prologue: tile (r=0,cs=0)
#endif

  for (int r = 0; r < 9; ++r) {
    const int kh = r / 3, kw = r - kh * 3;
    const _Float16* pB[4];
    bool vB[4];
    #pragma unroll
    for (int t = 0; t < 4; ++t) {
      int yy = oy[t] + kh - PAD, xx = ox[t] + kw - PAD;
      vB[t] = cv[t] && yy >= 0 && yy < H && xx >= 0 && xx < W;
      int yc = vB[t] ? yy : 0, xc = vB[t] ? xx : 0;
      pB[t] = base[t] + ((size_t)yc * W + xc) * CIN + lh * 16;
    }
    for (int cs = 0; cs < CIN; cs += 32) {
#if USE_TDM
      if (wave == 0) __builtin_amdgcn_s_wait_tensorcnt(0);   // current tile landed
      __syncthreads();   // LDS fence + barrier: tile visible; old buffer reads retired
      {
        int csn = cs + 32, rn = r;
        if (csn == CIN) { csn = 0; rn = r + 1; }
        if (wave == 0 && rn < 9)                             // pipeline next tile
          tdm_load_tile_2d(wblk + (size_t)rn * CIN + csn,
                           ldsA + (unsigned)(buf ^ 1) * 4096u, 32, 64, K9);
      }
#else
      {
        int row = tid >> 1, half = tid & 1;
        const _Float16* gA = wblk + (size_t)r * CIN + cs;
        *(v8h*)&As[buf][row][half * 16]     = *(const v8h*)(gA + (size_t)row * K9 + half * 16);
        *(v8h*)&As[buf][row][half * 16 + 8] = *(const v8h*)(gA + (size_t)row * K9 + half * 16 + 8);
      }
      __syncthreads();
#endif
      union { v16h v; v8h h[2]; } Af;
      Af.h[0] = *(const v8h*)&As[buf][mRow][lh * 16];
      Af.h[1] = *(const v8h*)&As[buf][mRow][lh * 16 + 8];
      v16h b0 = {}, b1 = {}, b2 = {}, b3 = {};
      if (vB[0]) b0 = *(const v16h*)(pB[0] + cs);
      if (vB[1]) b1 = *(const v16h*)(pB[1] + cs);
      if (vB[2]) b2 = *(const v16h*)(pB[2] + cs);
      if (vB[3]) b3 = *(const v16h*)(pB[3] + cs);
      acc0 = wmma_f16(Af.v, b0, acc0);
      acc1 = wmma_f16(Af.v, b1, acc1);
      acc2 = wmma_f16(Af.v, b2, acc2);
      acc3 = wmma_f16(Af.v, b3, acc3);
#if USE_TDM
      buf ^= 1;
#else
      __syncthreads();
#endif
    }
  }

  if constexpr (POOL) {
    int p = blockIdx.x * 16 + lr;
    if (p < NP) {
      int b = blockIdx.y;
      int poY = p / WO, poX = p - poY * WO;
      v8h o8;
      #pragma unroll
      for (int j = 0; j < 8; ++j) {
        int co = mBase + wave * 16 + lh * 8 + j;
        float v = fmaxf(fmaxf(acc0[j], acc1[j]), fmaxf(acc2[j], acc3[j])) + bias[co];
        o8[j] = (_Float16)fmaxf(v, 0.f);
      }
      *(v8h*)&out[(((size_t)b * HO + poY) * WO + poX) * COUT + mBase + wave * 16 + lh * 8] = o8;
    }
  } else {
    #pragma unroll
    for (int t = 0; t < 4; ++t) {
      v8f a = (t == 0) ? acc0 : (t == 1) ? acc1 : (t == 2) ? acc2 : acc3;
      int g = blockIdx.x * 64 + t * 16 + lr;
      if (g < B * NP) {
        v8h o8;
        #pragma unroll
        for (int j = 0; j < 8; ++j) {
          int co = mBase + wave * 16 + lh * 8 + j;
          o8[j] = (_Float16)fmaxf(a[j] + bias[co], 0.f);
        }
        *(v8h*)&out[(size_t)g * COUT + mBase + wave * 16 + lh * 8] = o8;
      }
    }
  }
}

// --------------------- 1x1 heads: 16x512 WMMA GEMM, direct fragments (no LDS)
__global__ __launch_bounds__(32) void heads_wmma_kernel(const _Float16* __restrict__ rp,
                                                        const _Float16* __restrict__ hw,
                                                        const float* __restrict__ hb,
                                                        float* __restrict__ out_cls,
                                                        float* __restrict__ out_reg, int B) {
  const int lane = threadIdx.x;
  const int lr = lane & 15, lh = lane >> 4;
  const _Float16* bp[4];
  bool bv[4];
  #pragma unroll
  for (int t = 0; t < 4; ++t) {
    int g = blockIdx.x * 64 + t * 16 + lr;
    bv[t] = g < B * 25;
    int gg = bv[t] ? g : 0;
    bp[t] = rp + (size_t)gg * 512 + lh * 16;
  }
  v8f acc0 = {}, acc1 = {}, acc2 = {}, acc3 = {};
  const _Float16* arow = hw + (size_t)lr * 512 + lh * 16;
  for (int ks = 0; ks < 512; ks += 32) {
    union { v16h v; v8h h[2]; } A;
    A.h[0] = *(const v8h*)(arow + ks);
    A.h[1] = *(const v8h*)(arow + ks + 8);
    v16h b0 = {}, b1 = {}, b2 = {}, b3 = {};
    if (bv[0]) b0 = *(const v16h*)(bp[0] + ks);
    if (bv[1]) b1 = *(const v16h*)(bp[1] + ks);
    if (bv[2]) b2 = *(const v16h*)(bp[2] + ks);
    if (bv[3]) b3 = *(const v16h*)(bp[3] + ks);
    acc0 = wmma_f16(A.v, b0, acc0);
    acc1 = wmma_f16(A.v, b1, acc1);
    acc2 = wmma_f16(A.v, b2, acc2);
    acc3 = wmma_f16(A.v, b3, acc3);
  }
  #pragma unroll
  for (int t = 0; t < 4; ++t) {
    v8f a = (t == 0) ? acc0 : (t == 1) ? acc1 : (t == 2) ? acc2 : acc3;
    int g = blockIdx.x * 64 + t * 16 + lr;
    if (g < B * 25) {
      int cb = g / 25, pix = g - cb * 25;
      #pragma unroll
      for (int j = 0; j < 8; ++j) {
        int row = lh * 8 + j;
        float v = a[j] + hb[row];
        if (row < 3)       out_cls[cb * 75 + row * 25 + pix] = 1.f / (1.f + __expf(-v));
        else if (row < 15) out_reg[cb * 300 + (row - 3) * 25 + pix] = v;
      }
    }
  }
}

// -------------------------------------------------------- FC: block per (b,o)
__global__ __launch_bounds__(256) void fc_kernel(const _Float16* __restrict__ feat,
                                                 const float* __restrict__ fw,
                                                 const float* __restrict__ fb,
                                                 float* __restrict__ logits) {
  const int K = 512 * 49;
  int b = blockIdx.x / 20, o = blockIdx.x - b * 20;
  float s = 0.f;
  for (int k = threadIdx.x; k < K; k += 256) {        // k = c*49 + pix (NCHW flat)
    int c = k / 49, pix = k - c * 49;
    s += (float)feat[((size_t)b * 49 + pix) * 512 + c] * fw[(size_t)o * K + k];
  }
  __shared__ float red[256];
  red[threadIdx.x] = s;
  __syncthreads();
  for (int st = 128; st > 0; st >>= 1) {
    if (threadIdx.x < st) red[threadIdx.x] += red[threadIdx.x + st];
    __syncthreads();
  }
  if (threadIdx.x == 0) logits[b * 20 + o] = red[0] + fb[o];
}

// ------------------------------------------- bbox deparam + greedy NMS (per b)
__global__ __launch_bounds__(128) void bbox_nms_kernel(const float* __restrict__ cls,
                                                       const float* __restrict__ reg,
                                                       float* __restrict__ bb_out,
                                                       float* __restrict__ keep_out) {
  const int b = blockIdx.x;
  const int t = threadIdx.x;
  __shared__ float bx[75][4];
  __shared__ float sb[75][4];
  __shared__ float sc[75];
  __shared__ int   ord[75];
  __shared__ int   keep[75];

  if (t < 75) {
    int gx = t / 15, rem = t - gx * 15, gy = rem / 3, k = rem - gy * 3;
    float aw = (k == 0) ? 56.f : (k == 1) ? 112.f : 224.f;
    const float* r = reg + b * 300 + t * 4;
    float cx = r[0] * aw + (float)gx;
    float cy = r[1] * aw + (float)gy;
    float w  = __expf(r[2]) * aw;
    float h  = __expf(r[3]) * aw;
    float x0 = cx - w * 0.5f, y0 = cy - h * 0.5f;
    float x1 = cx + w * 0.5f, y1 = cy + h * 0.5f;
    bx[t][0] = x0; bx[t][1] = y0; bx[t][2] = x1; bx[t][3] = y1;
    bb_out[b * 300 + t * 4 + 0] = x0;
    bb_out[b * 300 + t * 4 + 1] = y0;
    bb_out[b * 300 + t * 4 + 2] = x1;
    bb_out[b * 300 + t * 4 + 3] = y1;
    sc[t] = cls[b * 75 + t];
  }
  __syncthreads();
  if (t < 75) {                       // stable rank == argsort(-scores)
    float st = sc[t];
    int rank = 0;
    for (int j = 0; j < 75; ++j) {
      float sj = sc[j];
      rank += (sj > st) || (sj == st && j < t);
    }
    ord[rank] = t;
  }
  __syncthreads();
  if (t < 75) {
    int o = ord[t];
    sb[t][0] = bx[o][0]; sb[t][1] = bx[o][1];
    sb[t][2] = bx[o][2]; sb[t][3] = bx[o][3];
    keep[t] = 1;
  }
  __syncthreads();
  for (int i = 0; i < 75; ++i) {      // greedy suppression in score order
    if (t < 75 && t > i && keep[i]) {
      float xA = fmaxf(sb[i][0], sb[t][0]);
      float yA = fmaxf(sb[i][1], sb[t][1]);
      float xB = fminf(sb[i][2], sb[t][2]);
      float yB = fminf(sb[i][3], sb[t][3]);
      float inter = fmaxf(0.f, xB - xA + 1.f) * fmaxf(0.f, yB - yA + 1.f);
      float areaA = (sb[i][2] - sb[i][0] + 1.f) * (sb[i][3] - sb[i][1] + 1.f);
      float areaB = (sb[t][2] - sb[t][0] + 1.f) * (sb[t][3] - sb[t][1] + 1.f);
      float iou = inter / areaA + areaB - inter;   // faithful missing-parens bug
      if (iou > 0.7f) keep[t] = 0;
    }
    __syncthreads();
  }
  if (t < 75) keep_out[b * 75 + ord[t]] = keep[t] ? 1.f : 0.f;
}

// ----------------------------------------------------------------------------
extern "C" void kernel_launch(void* const* d_in, const int* in_sizes, int n_in,
                              void* d_out, int out_size, void* d_ws, size_t ws_size,
                              hipStream_t stream) {
  (void)in_sizes; (void)n_in; (void)out_size; (void)ws_size;
  const int B = 32;
  const float* x     = (const float*)d_in[0];
  const float* fw0   = (const float*)d_in[1];
  const float* fb0   = (const float*)d_in[2];
  const float* fw1   = (const float*)d_in[3];
  const float* fb1   = (const float*)d_in[4];
  const float* fw2   = (const float*)d_in[5];
  const float* fb2   = (const float*)d_in[6];
  const float* fw3   = (const float*)d_in[7];
  const float* fb3   = (const float*)d_in[8];
  const float* fw4   = (const float*)d_in[9];
  const float* fb4   = (const float*)d_in[10];
  const float* rpw_w = (const float*)d_in[11];
  const float* rpw_b = (const float*)d_in[12];
  const float* cls_w = (const float*)d_in[13];
  const float* cls_b = (const float*)d_in[14];
  const float* reg_w = (const float*)d_in[15];
  const float* reg_b = (const float*)d_in[16];
  const float* fc_w  = (const float*)d_in[17];
  const float* fc_b  = (const float*)d_in[18];
  float* out = (float*)d_out;

  // d_out layout (floats): logits[640] cls[2400] reg[9600] bboxes[9600] keep[2400]
  float* o_logits = out;
  float* o_cls    = out + 640;
  float* o_reg    = out + 3040;
  float* o_bbox   = out + 12640;
  float* o_keep   = out + 22240;

  char* ws = (char*)d_ws;
  size_t off = 0;
  auto carve = [&](size_t bytes) -> void* {
    void* p = ws + off;
    off = (off + bytes + 255) & ~(size_t)255;
    return p;
  };
  // ping-pong NHWC activation regions (f16): r0 = {x16, a2, a4}, r1 = {a1, a3}
  _Float16* r0    = (_Float16*)carve((size_t)12845056 * 2);
  _Float16* r1    = (_Float16*)carve((size_t)25690112 * 2);
  _Float16* feat  = (_Float16*)carve((size_t)802816 * 2);
  _Float16* rp512 = (_Float16*)carve((size_t)409600 * 2);
  _Float16* w0 = (_Float16*)carve((size_t)64  * 32   * 2);
  _Float16* w1 = (_Float16*)carve((size_t)128 * 576  * 2);
  _Float16* w2 = (_Float16*)carve((size_t)256 * 1152 * 2);
  _Float16* w3 = (_Float16*)carve((size_t)512 * 2304 * 2);
  _Float16* w4 = (_Float16*)carve((size_t)512 * 4608 * 2);
  _Float16* wr = (_Float16*)carve((size_t)512 * 4608 * 2);
  _Float16* wh = (_Float16*)carve((size_t)16  * 512  * 2);
  float*    hb = (float*)carve(16 * 4);

  // conversions / packing
  {
    int n = B * 3 * 224 * 224;
    nchw_to_nhwc_f16_kernel<<<(n + 255) / 256, 256, 0, stream>>>(x, r0, B, 3, 50176);
  }
  pack_weights_kernel<<<(64 * 32 + 255) / 256, 256, 0, stream>>>(fw0, w0, 64, 27, 32);
  pack_weights_cl_kernel<<<(128 * 576  + 255) / 256, 256, 0, stream>>>(fw1,   w1, 128, 64);
  pack_weights_cl_kernel<<<(256 * 1152 + 255) / 256, 256, 0, stream>>>(fw2,   w2, 256, 128);
  pack_weights_cl_kernel<<<(512 * 2304 + 255) / 256, 256, 0, stream>>>(fw3,   w3, 512, 256);
  pack_weights_cl_kernel<<<(512 * 4608 + 255) / 256, 256, 0, stream>>>(fw4,   w4, 512, 512);
  pack_weights_cl_kernel<<<(512 * 4608 + 255) / 256, 256, 0, stream>>>(rpw_w, wr, 512, 512);
  pack_heads_kernel<<<32, 256, 0, stream>>>(cls_w, cls_b, reg_w, reg_b, wh, hb);

  // backbone (NHWC, conv3x3 pad1 + relu + maxpool2 fused, WMMA implicit GEMM)
  conv0_wmma_kernel<<<dim3(784, 32, 1), 128, 0, stream>>>(r0, w0, fb0, r1, B);
  conv_cl_wmma_kernel<64,  128, 112, 112, 1, true><<<dim3(196, 32, 2), 128, 0, stream>>>(r1, w1, fb1, r0, B);
  conv_cl_wmma_kernel<128, 256, 56,  56,  1, true><<<dim3(49,  32, 4), 128, 0, stream>>>(r0, w2, fb2, r1, B);
  conv_cl_wmma_kernel<256, 512, 28,  28,  1, true><<<dim3(13,  32, 8), 128, 0, stream>>>(r1, w3, fb3, r0, B);
  conv_cl_wmma_kernel<512, 512, 14,  14,  1, true><<<dim3(4,   32, 8), 128, 0, stream>>>(r0, w4, fb4, feat, B);
  // RPN conv 3x3 valid + relu
  conv_cl_wmma_kernel<512, 512, 7,   7,   0, false><<<dim3(13, 1, 8), 128, 0, stream>>>(feat, wr, rpw_b, rp512, B);
  // heads (cls sigmoid + reg) as one stacked WMMA GEMM
  heads_wmma_kernel<<<13, 32, 0, stream>>>(rp512, wh, hb, o_cls, o_reg, B);
  // classifier
  fc_kernel<<<B * 20, 256, 0, stream>>>(feat, fc_w, fc_b, o_logits);
  // bbox deparam + NMS
  bbox_nms_kernel<<<B, 128, 0, stream>>>(o_cls, o_reg, o_bbox, o_keep);
}
